// CausalAttention_9569187136107
// MI455X (gfx1250) — matrix operand order repaired and checked by reading, not compiled, not run on hardware
//
#include <hip/hip_runtime.h>

// ---------------------------------------------------------------------------
// Causal attention forward for gfx1250 (MI455X), built from f16 WMMA GEMMs.
// Pipeline: f32->f16 convert | QKV projections | QK^T (causal-skip) |
//           row softmax (in-place) | P*V -> f32 out.
// All matrix math uses v_wmma_f32_16x16x32_f16 with f32 accumulation.
// Wave tile 64x64: 16 WMMAs per 8 fragment loads (2.0 wmma/frag).
// Leading dimensions are template constants so every fragment load uses the
// 24-bit immediate instruction offset off just two base pointers -> no VALU
// address math in the loop, no WMMA->VALU hazard NOPs.
// ---------------------------------------------------------------------------

typedef __attribute__((ext_vector_type(16))) _Float16 v16h;
typedef __attribute__((ext_vector_type(8)))  _Float16 v8h;
typedef __attribute__((ext_vector_type(8)))  float    v8f;

union HV16 { v16h v; v8h h[2]; };

__device__ __forceinline__ v8f wmma_f16(v16h a, v16h b, v8f c) {
  // (neg_a, A, neg_b, B, c_mod, C, reuse_a, reuse_b)
  return __builtin_amdgcn_wmma_f32_16x16x32_f16(false, a, false, b,
                                                (short)0, c, false, false);
}

// A fragment: 16x32 (MxK) f16, ISA 7.12.2:
//   lanes 0-15 : row M=lane,    VGPR0-3 = K 0..7,  VGPR4-7 = K 16..23
//   lanes 16-31: row M=lane-16, VGPR0-3 = K 8..15, VGPR4-7 = K 24..31
// p already includes the per-lane row and K sub-offset; OFF is a compile-time
// element offset (i*16*LDA).
template<int OFF>
__device__ __forceinline__ v16h load_a_frag(const _Float16* p) {
  HV16 f;
  f.h[0] = *(const v8h*)(p + OFF);
  f.h[1] = *(const v8h*)(p + OFF + 16);
  return f.v;
}

// B fragment: 32x16 (KxN) f16, operand stored row-major [N,K] so column n is
// K-contiguous: lanes 0-15 col N=lane (K 0..7, 16..23), lanes 16-31 col N=lane-16
// (K 8..15, 24..31). p includes per-lane column and K sub-offset.
template<int OFF>
__device__ __forceinline__ v16h load_b_frag(const _Float16* p) {
  HV16 f;
  f.h[0] = *(const v8h*)(p + OFF);
  f.h[1] = *(const v8h*)(p + OFF + 16);
  return f.v;
}

// ---------------------------------------------------------------------------
// Generic WMMA GEMM: C[M,N] = A[M,K] * B^T  (B stored row-major [N,K]).
// Workgroup tile 128x256: 8 waves of 64x64. blockIdx.z = batch.
// STORE_MODE: 0 = f16 row-major C, 1 = f16 transposed C (C[n,m], LDC = M-dim),
//             2 = f32 row-major C.
// CAUSAL_SKIP: skip WG tiles strictly above the diagonal.
// KBOUND: bound K loop at m0+128 (P*V causal K truncation).
// ---------------------------------------------------------------------------
template<int LDA, int LDB, int LDC, int STORE_MODE, bool CAUSAL_SKIP, bool KBOUND>
__global__ __launch_bounds__(256)
void gemm_wmma_k(const _Float16* __restrict__ A, const _Float16* __restrict__ Bm,
                 void* __restrict__ Cp, int K, long sA, long sB, long sC)
{
  const int bz = blockIdx.z;
  const _Float16* Ab = A  + (size_t)bz * sA;
  const _Float16* Bb = Bm + (size_t)bz * sB;
  const int m0 = blockIdx.x * 128;
  const int n0 = blockIdx.y * 256;
  if (CAUSAL_SKIP && (n0 > m0 + 127)) return;  // tile entirely above diagonal

  const int wave = threadIdx.x >> 5;
  const int lane = threadIdx.x & 31;
  const int row0 = m0 + (wave & 1) * 64;        // 2 waves along M
  const int col0 = n0 + (wave >> 1) * 64;       // 4 waves along N

  // Two live base pointers; all fragment addressing is immediate offsets.
  const _Float16* pa = Ab + (size_t)(row0 + (lane & 15)) * LDA + ((lane >> 4) << 3);
  const _Float16* pb = Bb + (size_t)(col0 + (lane & 15)) * LDB + ((lane >> 4) << 3);

  v8f acc[4][4] = {};

  int kend = K;
  if (KBOUND) { const int kb2 = m0 + 128; kend = (kb2 < K) ? kb2 : K; }

  for (int kb = 0; kb < kend; kb += 32) {
    v16h a0 = load_a_frag<0 * 16 * LDA>(pa);
    v16h a1 = load_a_frag<1 * 16 * LDA>(pa);
    v16h a2 = load_a_frag<2 * 16 * LDA>(pa);
    v16h a3 = load_a_frag<3 * 16 * LDA>(pa);
    v16h b0 = load_b_frag<0 * 16 * LDB>(pb);
    v16h b1 = load_b_frag<1 * 16 * LDB>(pb);
    v16h b2 = load_b_frag<2 * 16 * LDB>(pb);
    v16h b3 = load_b_frag<3 * 16 * LDB>(pb);
    pa += 32;
    pb += 32;

    acc[0][0] = wmma_f16(a0, b0, acc[0][0]);
    acc[0][1] = wmma_f16(a0, b1, acc[0][1]);
    acc[0][2] = wmma_f16(a0, b2, acc[0][2]);
    acc[0][3] = wmma_f16(a0, b3, acc[0][3]);
    acc[1][0] = wmma_f16(a1, b0, acc[1][0]);
    acc[1][1] = wmma_f16(a1, b1, acc[1][1]);
    acc[1][2] = wmma_f16(a1, b2, acc[1][2]);
    acc[1][3] = wmma_f16(a1, b3, acc[1][3]);
    acc[2][0] = wmma_f16(a2, b0, acc[2][0]);
    acc[2][1] = wmma_f16(a2, b1, acc[2][1]);
    acc[2][2] = wmma_f16(a2, b2, acc[2][2]);
    acc[2][3] = wmma_f16(a2, b3, acc[2][3]);
    acc[3][0] = wmma_f16(a3, b0, acc[3][0]);
    acc[3][1] = wmma_f16(a3, b1, acc[3][1]);
    acc[3][2] = wmma_f16(a3, b2, acc[3][2]);
    acc[3][3] = wmma_f16(a3, b3, acc[3][3]);
  }

  // C/D layout (ISA 7.12.2): VGPR r, lane: M = r + 8*(lane>=16), N = lane&15
  const int mrb = (lane >> 4) << 3;
  const int nc  = lane & 15;
  #pragma unroll
  for (int i = 0; i < 4; ++i) {
    #pragma unroll
    for (int j = 0; j < 4; ++j) {
      const int mb = row0 + i * 16 + mrb;
      const int nn = col0 + j * 16 + nc;
      if (STORE_MODE == 0) {
        _Float16* Cc = (_Float16*)Cp + (size_t)bz * sC;
        #pragma unroll
        for (int r = 0; r < 8; ++r)
          Cc[(size_t)(mb + r) * LDC + nn] = (_Float16)acc[i][j][r];
      } else if (STORE_MODE == 1) {
        _Float16* Cc = (_Float16*)Cp + (size_t)bz * sC;
        v8h t;
        #pragma unroll
        for (int r = 0; r < 8; ++r) t[r] = (_Float16)acc[i][j][r];
        *(v8h*)(Cc + (size_t)nn * LDC + mb) = t;   // 8 contiguous along M
      } else {
        float* Cc = (float*)Cp + (size_t)bz * sC;
        #pragma unroll
        for (int r = 0; r < 8; ++r)
          Cc[(size_t)(mb + r) * LDC + nn] = acc[i][j][r];
      }
    }
  }
}

// ---------------------------------------------------------------------------
// Causal row softmax, in-place over f16 score rows. One block per (row, batch).
// Writes zeros into the masked region j > row so P*V needs no masking.
// ---------------------------------------------------------------------------
__global__ __launch_bounds__(256)
void softmax_causal_k(_Float16* __restrict__ SP, int n, float scale)
{
  const int row = blockIdx.x;
  const int bz  = blockIdx.y;
  _Float16* s = SP + ((size_t)bz * n + row) * (size_t)n;
  const int len  = row + 1;
  const int tid  = threadIdx.x;
  const int lane = tid & 31;
  const int wave = tid >> 5;
  __shared__ float red[8];

  float m = -3.0e38f;
  for (int j = tid; j < len; j += 256)
    m = fmaxf(m, (float)s[j] * scale);
  #pragma unroll
  for (int o = 16; o > 0; o >>= 1) m = fmaxf(m, __shfl_xor(m, o, 32));
  if (lane == 0) red[wave] = m;
  __syncthreads();
  float mrow = -3.0e38f;
  #pragma unroll
  for (int w = 0; w < 8; ++w) mrow = fmaxf(mrow, red[w]);
  __syncthreads();

  float ssum = 0.0f;
  for (int j = tid; j < len; j += 256)
    ssum += __expf((float)s[j] * scale - mrow);
  #pragma unroll
  for (int o = 16; o > 0; o >>= 1) ssum += __shfl_xor(ssum, o, 32);
  if (lane == 0) red[wave] = ssum;
  __syncthreads();
  float sum = 0.0f;
  #pragma unroll
  for (int w = 0; w < 8; ++w) sum += red[w];
  const float rs = 1.0f / sum;
  __syncthreads();

  for (int j = tid; j < n; j += 256) {
    float v = 0.0f;
    if (j < len) v = __expf((float)s[j] * scale - mrow) * rs;
    s[j] = (_Float16)v;
  }
}

__global__ __launch_bounds__(256)
void cvt_f32_f16_k(const float* __restrict__ in, _Float16* __restrict__ out, int n)
{
  const int i = blockIdx.x * 256 + threadIdx.x;
  if (i < n) out[i] = (_Float16)in[i];
}

// ---------------------------------------------------------------------------
extern "C" void kernel_launch(void* const* d_in, const int* in_sizes, int n_in,
                              void* d_out, int out_size, void* d_ws, size_t ws_size,
                              hipStream_t stream)
{
  (void)in_sizes; (void)n_in; (void)out_size; (void)ws_size;
  const float* x  = (const float*)d_in[0];
  const float* Wq = (const float*)d_in[1];
  const float* Wk = (const float*)d_in[2];
  const float* Wv = (const float*)d_in[3];

  const int Bn = 4, S = 4096, DI = 1024, DO = 1024;
  const size_t nx  = (size_t)Bn * S * DI;   // 16,777,216
  const size_t nw  = (size_t)DO * DI;       //  1,048,576
  const size_t nq  = (size_t)Bn * S * DO;   // 16,777,216
  const size_t nsc = (size_t)Bn * S * S;    // 67,108,864

  // Workspace layout (f16 elements): ~275 MB total
  _Float16* ws  = (_Float16*)d_ws;
  size_t off = 0;
  _Float16* xb  = ws + off; off += nx;
  _Float16* wqb = ws + off; off += nw;
  _Float16* wkb = ws + off; off += nw;
  _Float16* wvb = ws + off; off += nw;
  _Float16* qb  = ws + off; off += nq;      // [B, S, DO] row-major
  _Float16* kb  = ws + off; off += nq;      // [B, S, DO] row-major
  _Float16* vtb = ws + off; off += nq;      // [B, DO, S] (transposed V)
  _Float16* sc  = ws + off; off += nsc;     // [B, S, S] scores -> P (in place)

  // 1) Convert inputs to f16
  cvt_f32_f16_k<<<dim3((unsigned)((nx + 255) / 256)), dim3(256), 0, stream>>>(x,  xb,  (int)nx);
  cvt_f32_f16_k<<<dim3((unsigned)((nw + 255) / 256)), dim3(256), 0, stream>>>(Wq, wqb, (int)nw);
  cvt_f32_f16_k<<<dim3((unsigned)((nw + 255) / 256)), dim3(256), 0, stream>>>(Wk, wkb, (int)nw);
  cvt_f32_f16_k<<<dim3((unsigned)((nw + 255) / 256)), dim3(256), 0, stream>>>(Wv, wvb, (int)nw);

  // 2) QKV projections: y = x @ W^T, per batch (z). v stored transposed.
  dim3 gq(S / 128, DO / 256, Bn);
  gemm_wmma_k<1024, 1024, 1024, 0, false, false><<<gq, 256, 0, stream>>>(
      xb, wqb, qb,  DI, (long)S * DI, 0L, (long)S * DO);
  gemm_wmma_k<1024, 1024, 1024, 0, false, false><<<gq, 256, 0, stream>>>(
      xb, wkb, kb,  DI, (long)S * DI, 0L, (long)S * DO);
  gemm_wmma_k<1024, 1024, 4096, 1, false, false><<<gq, 256, 0, stream>>>(
      xb, wvb, vtb, DI, (long)S * DI, 0L, (long)DO * S);

  // 3) scores = q @ k^T  (skip tiles above the diagonal)
  dim3 gs(S / 128, S / 256, Bn);
  gemm_wmma_k<1024, 1024, 4096, 0, true, false><<<gs, 256, 0, stream>>>(
      qb, kb, sc, DO, (long)S * DO, (long)S * DO, (long)S * S);

  // 4) causal softmax in place (scale = 1/sqrt(1024))
  softmax_causal_k<<<dim3(S, Bn), dim3(256), 0, stream>>>(sc, S, 0.03125f);

  // 5) out = P @ V  (V^T as the [N,K] B-operand; K bounded at diagonal)
  dim3 gp(S / 128, DO / 256, Bn);
  gemm_wmma_k<4096, 4096, 1024, 2, false, true><<<gp, 256, 0, stream>>>(
      sc, vtb, d_out, S, (long)S * S, (long)DO * S, (long)S * DO);
}
